// Block_30520037605534
// MI455X (gfx1250) — matrix operand order repaired and checked by reading, not compiled, not run on hardware
//
#include <hip/hip_runtime.h>
#include <hip/hip_bf16.h>

typedef __attribute__((ext_vector_type(16))) __bf16 v16bf;
typedef __attribute__((ext_vector_type(8)))  __bf16 v8bf;
typedef __attribute__((ext_vector_type(4)))  __bf16 v4bf;
typedef __attribute__((ext_vector_type(2)))  __bf16 v2bf;
typedef __attribute__((ext_vector_type(8)))  float  v8f;

static constexpr int Bc  = 2;
static constexpr int Tc  = 2048;
static constexpr int Cc  = 1024;
static constexpr int Hc  = 16;
static constexpr int HSc = 64;

struct u32x2 { unsigned u[2]; };
struct u32x4 { unsigned u[4]; };
struct u32x8 { unsigned u[8]; };

__device__ __forceinline__ __bf16 f2bf(float f) {
  unsigned u = __builtin_bit_cast(unsigned, f);
  u += 0x7FFFu + ((u >> 16) & 1u);               // round-to-nearest-even
  unsigned short h = (unsigned short)(u >> 16);
  return __builtin_bit_cast(__bf16, h);
}

// pack two floats as adjacent bf16 values in one dword (elem0 = low half)
__device__ __forceinline__ unsigned pack2bf(float a, float b) {
#if __has_builtin(__builtin_amdgcn_cvt_pk_bf16_f32)
  return __builtin_bit_cast(unsigned, __builtin_amdgcn_cvt_pk_bf16_f32(a, b));
#else
  unsigned ua = (unsigned)__builtin_bit_cast(unsigned short, f2bf(a));
  unsigned ub = (unsigned)__builtin_bit_cast(unsigned short, f2bf(b));
  return ua | (ub << 16);
#endif
}

// lane <-> lane^16 swap. v_permlanex16 is pure VALU (no LDS round trip).
__device__ __forceinline__ unsigned swap16u(unsigned v) {
#if __has_builtin(__builtin_amdgcn_permlanex16)
  return __builtin_amdgcn_permlanex16(v, v, 0x76543210u, 0xfedcba98u, false,
                                      false);
#else
  return __shfl_xor(v, 16);
#endif
}
__device__ __forceinline__ float swap16f(float v) {
  return __builtin_bit_cast(float, swap16u(__builtin_bit_cast(unsigned, v)));
}

struct bfpair { v8bf lo, hi; };
// A fragment: 16 contiguous elements at p
__device__ __forceinline__ v16bf ldfragA(const __bf16* p) {
  bfpair s{ *(const v8bf*)p, *(const v8bf*)(p + 8) };
  return __builtin_bit_cast(v16bf, s);
}
// B fragment: 16 contiguous elements at p
__device__ __forceinline__ v16bf ldfragB(const __bf16* p) {
  bfpair s{ *(const v8bf*)p, *(const v8bf*)(p + 8) };
  return __builtin_bit_cast(v16bf, s);
}

__device__ __forceinline__ v8f wmma_bf16(v16bf a, v16bf b, v8f c) {
  return __builtin_amdgcn_wmma_f32_16x16x32_bf16(false, a, false, b, (short)0, c,
                                                 false, false);
}

// CDNA5 async copy: 16B global -> LDS, tracked by ASYNCcnt
__device__ __forceinline__ void async_cp16(void* lds, const void* g) {
  asm volatile("global_load_async_to_lds_b128 %0, %1, off"
               :: "v"((unsigned)(uintptr_t)lds),
                  "v"((unsigned long long)(uintptr_t)g)
               : "memory");
}
__device__ __forceinline__ void wait_async0() {
  asm volatile("s_wait_asynccnt 0" ::: "memory");
}

// ---------------------------------------------------------------- convert
__global__ void f32_to_bf16_kernel(const float* __restrict__ in,
                                   __bf16* __restrict__ out, int n) {
  int i4 = (blockIdx.x * 256 + threadIdx.x) * 4;
  if (i4 < n) {
    float4 f = *(const float4*)(in + i4);
    u32x2 p{ { pack2bf(f.x, f.y), pack2bf(f.z, f.w) } };
    *(v4bf*)(out + i4) = __builtin_bit_cast(v4bf, p);
  }
}

// ---------------------------------------------------------------- layernorm
__global__ __launch_bounds__(256) void ln_bf16_kernel(
    const float* __restrict__ x, const float* __restrict__ g,
    const float* __restrict__ bta, __bf16* __restrict__ out) {
  const int row  = blockIdx.x;
  const int tid  = threadIdx.x;
  const int lane = tid & 31;
  const int w    = tid >> 5;
  const float* xr = x + (size_t)row * Cc;
  float s = 0.f, s2 = 0.f;
  for (int i = tid; i < Cc; i += 256) { float v = xr[i]; s += v; s2 += v * v; }
#pragma unroll
  for (int off = 16; off > 0; off >>= 1) {
    s  += __shfl_xor(s, off);
    s2 += __shfl_xor(s2, off);
  }
  __shared__ float sm[8], sq[8];
  if (lane == 0) { sm[w] = s; sq[w] = s2; }
  __syncthreads();
  float ts = 0.f, tq = 0.f;
#pragma unroll
  for (int j = 0; j < 8; ++j) { ts += sm[j]; tq += sq[j]; }
  const float mu   = ts * (1.0f / Cc);
  const float var  = tq * (1.0f / Cc) - mu * mu;
  const float rstd = rsqrtf(var + 1e-5f);
  for (int i = tid; i < Cc; i += 256)
    out[(size_t)row * Cc + i] = f2bf((xr[i] - mu) * rstd * g[i] + bta[i]);
}

// ---------------------------------------------------------------- WMMA GEMM
// Block tile 128x64, 8 waves (4x2), each wave 32x32 (4 WMMAs/K-step).
// A tiles: CDNA5 async global->LDS copies (ASYNCcnt), double-buffered.
// B tiles: register prefetch + transposed LDS store so fragments are b128 loads.
template <bool BIAS, bool RES, bool RELU, bool BF16OUT, bool TRANS, bool SCALE>
__global__ __launch_bounds__(256) void gemm_wmma(
    const __bf16* __restrict__ A, const __bf16* __restrict__ Bm,
    const float* __restrict__ bias, const float* __restrict__ resid,
    void* __restrict__ Out, int K, int ldA, int ldB, int ldO, float oscale,
    int a_div, long long a_stride, int b_mod, long long b_stride,
    long long o_stride) {
  __shared__ __align__(16) __bf16 As[2][128][32];
  __shared__ __align__(16) __bf16 Bt[2][64][40];   // [n][k], padded rows (80B)

  const int tid  = threadIdx.x;
  const int lane = tid & 31;
  const int w    = tid >> 5;
  const int wm   = w >> 1;           // 0..3 -> 32-row strip
  const int wn   = w & 1;            // 0..1 -> 32-col strip
  const int hi   = (lane >> 4) & 1;
  const int ln16 = lane & 15;

  const int m0 = blockIdx.x * 128;
  const int n0 = blockIdx.y * 64;
  const int z  = blockIdx.z;
  A  += (long long)(z / a_div) * a_stride;
  Bm += (long long)(z % b_mod) * b_stride;

  const int ar = tid >> 2, ac = (tid & 3) << 3;   // A rows ar, ar+64
  const int br = tid >> 3, bc = (tid & 7) << 3;   // B row br, cols bc..bc+7
  const __bf16* Ap0 = A + (size_t)(m0 + ar) * ldA + ac;
  const __bf16* Ap1 = A + (size_t)(m0 + ar + 64) * ldA + ac;
  const __bf16* Bp  = Bm + (size_t)br * ldB + n0 + bc;

  // kick off async copies for buffer 0
  async_cp16(&As[0][ar][ac],      Ap0);
  async_cp16(&As[0][ar + 64][ac], Ap1);
  v8bf rb = *(const v8bf*)Bp;

  v8f acc[2][2];
#pragma unroll
  for (int mi = 0; mi < 2; ++mi)
#pragma unroll
    for (int ni = 0; ni < 2; ++ni)
      acc[mi][ni] = (v8f){0.f,0.f,0.f,0.f,0.f,0.f,0.f,0.f};

  int buf = 0;
  for (int k0 = 0; k0 < K; k0 += 32) {
#pragma unroll
    for (int e = 0; e < 8; ++e) Bt[buf][bc + e][br] = rb[e];
    wait_async0();                            // our A copies into buf are done
    __syncthreads();                          // everyone's copies/stores visible
    if (k0 + 32 < K) {                        // stage next K-slab
      async_cp16(&As[buf ^ 1][ar][ac],      Ap0 + k0 + 32);
      async_cp16(&As[buf ^ 1][ar + 64][ac], Ap1 + k0 + 32);
      rb = *(const v8bf*)(Bp + (size_t)(k0 + 32) * ldB);
    }
    v16bf af[2], bf[2];
#pragma unroll
    for (int s = 0; s < 2; ++s) {
      af[s] = ldfragA(&As[buf][wm * 32 + s * 16 + ln16][hi * 16]);
      bf[s] = ldfragB(&Bt[buf][wn * 32 + s * 16 + ln16][hi * 16]);
    }
#pragma unroll
    for (int mi = 0; mi < 2; ++mi)
#pragma unroll
      for (int ni = 0; ni < 2; ++ni)
        acc[mi][ni] = wmma_bf16(af[mi], bf[ni], acc[mi][ni]);
    buf ^= 1;
  }

#pragma unroll
  for (int mi = 0; mi < 2; ++mi)
#pragma unroll
    for (int ni = 0; ni < 2; ++ni) {
      const int col = n0 + wn * 32 + ni * 16 + ln16;
      float vv[8];
#pragma unroll
      for (int i = 0; i < 8; ++i) {
        const int row = m0 + wm * 32 + mi * 16 + i + 8 * hi;
        float v = acc[mi][ni][i];
        if (SCALE) v *= oscale;
        if (BIAS)  v += bias[col];
        if (RELU)  v = fmaxf(v, 0.f);
        if (RES)   v += resid[(size_t)row * ldO + col];
        vv[i] = v;
      }
      if (TRANS) {
        // col-major output: the 8 per-lane rows are contiguous -> 16B store
        const int row0 = m0 + wm * 32 + mi * 16 + 8 * hi;
        u32x4 pk{ { pack2bf(vv[0], vv[1]), pack2bf(vv[2], vv[3]),
                    pack2bf(vv[4], vv[5]), pack2bf(vv[6], vv[7]) } };
        *(v8bf*)((__bf16*)Out + (size_t)z * o_stride + (size_t)col * ldO +
                 row0) = __builtin_bit_cast(v8bf, pk);
      } else {
#pragma unroll
        for (int i = 0; i < 8; ++i) {
          const int row = m0 + wm * 32 + mi * 16 + i + 8 * hi;
          const size_t oidx = (size_t)z * o_stride + (size_t)row * ldO + col;
          if (BF16OUT) ((__bf16*)Out)[oidx] = f2bf(vv[i]);
          else         ((float*) Out)[oidx] = vv[i];
        }
      }
    }
}

// ---------------------------------------------------------------- causal attention
// Transposed formulation: each lane owns one query column.
//  S^T = K . Q^T   (A = K tile, B = Q^T)  -> C layout: row=key, col=query
//  O^T = V^T . P^T (A = V^T tile, B = P^T)
// Reductions: in-register trees + one permlanex16. P^T half-swap done on
// packed bf16 pairs (8 swaps instead of 16). Q pre-scaled by 1/8.
template <bool MASKED>
__device__ __forceinline__ void attn_chunk(
    int s0, int qb, int ln16, int hi, const __bf16* __restrict__ kg,
    const __bf16* __restrict__ vg, const v16bf& bq0, const v16bf& bq1,
    v8f acc[4], float& m, float& l) {
  v8f sc0 = (v8f){0.f,0.f,0.f,0.f,0.f,0.f,0.f,0.f};
  v8f sc1 = sc0;
#pragma unroll
  for (int j = 0; j < 2; ++j) {
    const __bf16* kr = kg + (size_t)(s0 + j * 16 + ln16) * HSc + hi * 16;
    const v16bf ak0 = ldfragA(kr);
    const v16bf ak1 = ldfragA(kr + 32);
    if (j == 0) {
      sc0 = wmma_bf16(ak0, bq0, sc0);
      sc0 = wmma_bf16(ak1, bq1, sc0);
    } else {
      sc1 = wmma_bf16(ak0, bq0, sc1);
      sc1 = wmma_bf16(ak1, bq1, sc1);
    }
  }

  float p0[8], p1[8];
#pragma unroll
  for (int i = 0; i < 8; ++i) {
    float a = sc0[i], b = sc1[i];
    if (MASKED) {
      const int qrow = qb + ln16;
      if (s0 + i + 8 * hi > qrow)      a = -__builtin_inff();
      if (s0 + 16 + i + 8 * hi > qrow) b = -__builtin_inff();
    }
    p0[i] = a; p1[i] = b;
  }
  // per-query max: in-register tree + one cross-half swap
  float mx = fmaxf(p0[0], p1[0]);
#pragma unroll
  for (int i = 1; i < 8; ++i) mx = fmaxf(mx, fmaxf(p0[i], p1[i]));
  mx = fmaxf(mx, swap16f(mx));
  const float mn    = fmaxf(m, mx);
  const float alpha = __expf(m - mn);
  m = mn;
  float sum = 0.f;
#pragma unroll
  for (int i = 0; i < 8; ++i) {
    p0[i] = __expf(p0[i] - mn);
    p1[i] = __expf(p1[i] - mn);
    sum += p0[i] + p1[i];
  }
  sum += swap16f(sum);
  l = l * alpha + sum;
#pragma unroll
  for (int t = 0; t < 4; ++t)
#pragma unroll
    for (int i = 0; i < 8; ++i) acc[t][i] *= alpha;

  // P^T B-fragment: element e is key 16*hi+e. Convert+pack to bf16 pairs,
  // swap halves on packed dwords, then select.
  u32x8 bpw;
#pragma unroll
  for (int i = 0; i < 4; ++i) {
    const unsigned c0 = pack2bf(p0[2 * i], p0[2 * i + 1]);
    const unsigned c1 = pack2bf(p1[2 * i], p1[2 * i + 1]);
    const unsigned x0 = swap16u(c0);
    const unsigned x1 = swap16u(c1);
    bpw.u[i]     = hi ? x1 : c0;
    bpw.u[4 + i] = hi ? c1 : x0;
  }
  const v16bf bp = __builtin_bit_cast(v16bf, bpw);

  // O^T += V^T . P^T
#pragma unroll
  for (int t = 0; t < 4; ++t) {
    const v16bf av = ldfragA(vg + (size_t)(t * 16 + ln16) * Tc + s0 + hi * 16);
    acc[t] = wmma_bf16(av, bp, acc[t]);
  }
}

__global__ __launch_bounds__(128) void attn_wmma_kernel(
    const __bf16* __restrict__ q, const __bf16* __restrict__ k,
    const __bf16* __restrict__ v, __bf16* __restrict__ o) {
  const int tid  = threadIdx.x;
  const int lane = tid & 31;
  const int w    = tid >> 5;
  const int hi   = (lane >> 4) & 1;
  const int ln16 = lane & 15;
  const int bh   = blockIdx.y;
  const int bidx = bh >> 4, hidx = bh & 15;
  const int qb   = blockIdx.x * 64 + w * 16;

  const __bf16* qg = q + (size_t)bh * Tc * HSc;
  const __bf16* kg = k + (size_t)bh * Tc * HSc;
  const __bf16* vg = v + (size_t)bh * Tc * HSc;   // [hs][T]

  // Q^T B-fragments: col = query (lane%16), K = d contiguous. Loaded once.
  const __bf16* qr = qg + (size_t)(qb + ln16) * HSc + hi * 16;
  const v16bf bq0 = ldfragB(qr);
  const v16bf bq1 = ldfragB(qr + 32);

  v8f acc[4];
#pragma unroll
  for (int t = 0; t < 4; ++t) acc[t] = (v8f){0.f,0.f,0.f,0.f,0.f,0.f,0.f,0.f};
  float m = -__builtin_inff(), l = 0.f;

  const int nfull = (qb + 1) >> 5;       // chunks with no masked element
  for (int c = 0; c < nfull; ++c)
    attn_chunk<false>(c << 5, qb, ln16, hi, kg, vg, bq0, bq1, acc, m, l);
  attn_chunk<true>(nfull << 5, qb, ln16, hi, kg, vg, bq0, bq1, acc, m, l);

  // write O: per lane, 8 consecutive d values per tile -> 16B packed stores
  const float rinv = 1.0f / l;
  __bf16* op = o + ((size_t)(bidx * Tc + qb + ln16)) * Cc + hidx * HSc;
#pragma unroll
  for (int t = 0; t < 4; ++t) {
    u32x4 pk{ { pack2bf(acc[t][0] * rinv, acc[t][1] * rinv),
                pack2bf(acc[t][2] * rinv, acc[t][3] * rinv),
                pack2bf(acc[t][4] * rinv, acc[t][5] * rinv),
                pack2bf(acc[t][6] * rinv, acc[t][7] * rinv) } };
    *(v8bf*)(op + t * 16 + 8 * hi) = __builtin_bit_cast(v8bf, pk);
  }
}

// ---------------------------------------------------------------- host
extern "C" void kernel_launch(void* const* d_in, const int* in_sizes, int n_in,
                              void* d_out, int out_size, void* d_ws, size_t ws_size,
                              hipStream_t stream) {
  const float* x     = (const float*)d_in[0];
  const float* Wq    = (const float*)d_in[1];
  const float* Wk    = (const float*)d_in[2];
  const float* Wv    = (const float*)d_in[3];
  const float* Wproj = (const float*)d_in[4];
  const float* bproj = (const float*)d_in[5];
  const float* g1    = (const float*)d_in[6];
  const float* b1    = (const float*)d_in[7];
  const float* g2    = (const float*)d_in[8];
  const float* b2    = (const float*)d_in[9];
  const float* Wff1  = (const float*)d_in[10];
  const float* bff1  = (const float*)d_in[11];
  const float* Wff2  = (const float*)d_in[12];
  const float* bff2  = (const float*)d_in[13];
  float* out = (float*)d_out;

  const int BT = Bc * Tc;  // 4096

  char* ws = (char*)d_ws;
  size_t off = 0;
  auto carve = [&](size_t elems) -> __bf16* {
    __bf16* p = (__bf16*)(ws + off);
    off = (off + elems * 2 + 255) & ~(size_t)255;
    return p;
  };
  __bf16* hbf  = carve((size_t)BT * Cc);        // LN1 output
  __bf16* wqb  = carve((size_t)Cc * Cc);
  __bf16* wkb  = carve((size_t)Cc * Cc);
  __bf16* wvb  = carve((size_t)Cc * Cc);
  __bf16* wpb  = carve((size_t)Cc * Cc);
  __bf16* wf1b = carve((size_t)Cc * 4 * Cc);
  __bf16* wf2b = carve((size_t)4 * Cc * Cc);
  __bf16* qbf  = carve((size_t)BT * Cc);        // [B,H,T,hs] (pre-scaled)
  __bf16* kbf  = carve((size_t)BT * Cc);        // [B,H,T,hs]
  __bf16* vbf  = carve((size_t)BT * Cc);        // [B,H,hs,T] (transposed)
  __bf16* obf  = carve((size_t)BT * Cc);        // attn out [B,T,C]
  __bf16* h2bf = carve((size_t)BT * Cc);        // LN2 output
  __bf16* ffab = carve((size_t)BT * 4 * Cc);    // relu(FFN1)

  auto conv = [&](const float* src, __bf16* dst, int n) {
    f32_to_bf16_kernel<<<(n / 4 + 255) / 256, 256, 0, stream>>>(src, dst, n);
  };
  conv(Wq,    wqb,  Cc * Cc);
  conv(Wk,    wkb,  Cc * Cc);
  conv(Wv,    wvb,  Cc * Cc);
  conv(Wproj, wpb,  Cc * Cc);
  conv(Wff1,  wf1b, Cc * 4 * Cc);
  conv(Wff2,  wf2b, 4 * Cc * Cc);

  // LN1: x -> hbf
  ln_bf16_kernel<<<BT, 256, 0, stream>>>(x, g1, b1, hbf);

  // QKV: per (b,h): [T,C] x [C,hs] -> q/k [T,hs], v [hs,T];  z = b*H+h
  {
    dim3 grid(Tc / 128, 1, Bc * Hc);
    const long long aS = (long long)Tc * Cc;   // per-b
    const long long bS = (long long)Cc * HSc;  // per-h
    const long long oS = (long long)Tc * HSc;  // per-z
    gemm_wmma<false, false, false, true, false, true><<<grid, 256, 0, stream>>>(
        hbf, wqb, nullptr, nullptr, qbf, Cc, Cc, HSc, HSc, 0.125f,
        Hc, aS, Hc, bS, oS);
    gemm_wmma<false, false, false, true, false, false><<<grid, 256, 0, stream>>>(
        hbf, wkb, nullptr, nullptr, kbf, Cc, Cc, HSc, HSc, 1.0f,
        Hc, aS, Hc, bS, oS);
    gemm_wmma<false, false, false, true, true, false><<<grid, 256, 0, stream>>>(
        hbf, wvb, nullptr, nullptr, vbf, Cc, Cc, HSc, Tc, 1.0f,
        Hc, aS, Hc, bS, oS);
  }

  // causal attention
  attn_wmma_kernel<<<dim3(Tc / 64, Bc * Hc), 128, 0, stream>>>(qbf, kbf, vbf, obf);

  // proj + bias + residual(x) -> d_out (x1, f32)
  gemm_wmma<true, true, false, false, false, false>
      <<<dim3(BT / 128, Cc / 64, 1), 256, 0, stream>>>(
          obf, wpb, bproj, x, out, Cc, Cc, Cc, Cc, 1.0f, 1, 0, 1, 0, 0);

  // LN2: x1 -> h2bf
  ln_bf16_kernel<<<BT, 256, 0, stream>>>(out, g2, b2, h2bf);

  // FFN1: relu(h2 @ Wff1 + bff1) -> ffab (bf16)
  gemm_wmma<true, false, true, true, false, false>
      <<<dim3(BT / 128, (4 * Cc) / 64, 1), 256, 0, stream>>>(
          h2bf, wf1b, bff1, nullptr, ffab, Cc, Cc, 4 * Cc, 4 * Cc, 1.0f,
          1, 0, 1, 0, 0);

  // FFN2: ffab @ Wff2 + bff2 + x1 -> d_out (f32, in-place residual)
  gemm_wmma<true, true, false, false, false, false>
      <<<dim3(BT / 128, Cc / 64, 1), 256, 0, stream>>>(
          ffab, wf2b, bff2, out, out, 4 * Cc, 4 * Cc, Cc, Cc, 1.0f,
          1, 0, 1, 0, 0);
}
